// GPT2DecoderLayer_32822140076226
// MI455X (gfx1250) — matrix operand order repaired and checked by reading, not compiled, not run on hardware
//
#include <hip/hip_runtime.h>

// ---------------------------------------------------------------------------
// GPT-2 decoder layer forward for MI455X (gfx1250), bf16 WMMA path.
//
// ~120 GFLOP of GEMM vs ~0.4GB traffic -> matrix-core bound. All GEMMs run
// through V_WMMA_F32_16X16X32_BF16 (fp32 accum). bf16 over fp16 because the
// reference multiplies V by raw masked scores (-1e9) which overflows fp16.
//
// New this round: tile staging uses GLOBAL_LOAD_ASYNC_TO_LDS_B128 (ASYNCcnt)
// with double-buffered LDS so the k+1 tile streams in while tile k's WMMAs
// execute -- the CDNA5 async copy path, no VGPR round-trip.
// ---------------------------------------------------------------------------

typedef __attribute__((ext_vector_type(16))) __bf16 v16bf;
typedef __attribute__((ext_vector_type(8)))  float  v8f;

constexpr int Bc  = 4;
constexpr int Sc  = 1024;
constexpr int Hc  = 16;
constexpr int DHc = 64;
constexpr int Dc  = 1024;
constexpr int DFFc = 4096;
constexpr int BSc = Bc * Sc;      // 4096 rows

// Epilogue kinds
constexpr int EPI_BF16      = 0;  // store bf16 C
constexpr int EPI_SCORES    = 1;  // scale, causal mask, store fp32 + bf16
constexpr int EPI_BIAS_F32  = 2;  // + bias, store fp32
constexpr int EPI_GELU_BF16 = 3;  // + bias, exact GELU, store bf16

union FragU { v16bf v; uint4 q[2]; };

// Generic pointer to LDS byte offset (addrspacecast p3->p0 keeps the LDS
// offset in the low 32 bits).
__device__ __forceinline__ unsigned lds_off(const void* p)
{
    return (unsigned)(unsigned long long)p;
}

// ---------------------------------------------------------------------------
// Generic bf16 GEMM: C[M,N] = A[M,K] (row-major) x B[K,N] (K-major/row-major)
// Workgroup: 256 threads = 8 waves. Tiles: BM x BN, wave tile WM x WN.
// Fragment layouts per CDNA5 ISA 7.12.2:
//   A frag: lane&15 = M row; lanes<16 hold K 0-7,16-23; lanes>=16 K 8-15,24-31
//   B frag: lane&15 = N col; lanes<16 hold K 0-7,16-23; lanes>=16 K 8-15,24-31
// ---------------------------------------------------------------------------
template<int BM, int BN, int WM, int WN, int EPI>
__global__ __launch_bounds__(256) void gemm_bf16_kernel(
    const __bf16* __restrict__ A, const __bf16* __restrict__ B,
    float* __restrict__ Cf, __bf16* __restrict__ Cb,
    int lda, int ldb, int ldc, int K, int Hh,
    long long aOffB, long long aOffH,
    long long bOffB, long long bOffH,
    long long cOffB, long long cOffH,
    const float* __restrict__ bias, float scale)
{
    constexpr int KT   = 32;
    constexpr int ASTR = KT + 8;      // LDS row stride (bf16), keeps 16B align
    constexpr int BSTR = BN + 8;
    constexpr int MF = WM / 16, NF = WN / 16;
    constexpr int WAVES_N = BN / WN;
    constexpr int AVECS = (BM * KT) / (256 * 8);
    constexpr int BVECS = (KT * BN) / (256 * 8);

    __shared__ __align__(16) __bf16 As[2][BM * ASTR];
    __shared__ __align__(16) __bf16 Bs[2][KT * BSTR];

    const int tid  = threadIdx.x;
    const int lane = tid & 31;
    const int wave = tid >> 5;
    const int wm = wave / WAVES_N;
    const int wn = wave % WAVES_N;

    const int z  = blockIdx.z;
    const int bb = z / Hh;
    const int hh = z % Hh;
    A += (long long)bb * aOffB + (long long)hh * aOffH;
    B += (long long)bb * bOffB + (long long)hh * bOffH;
    const long long cOff = (long long)bb * cOffB + (long long)hh * cOffH;

    const int m0 = blockIdx.y * BM;
    const int n0 = blockIdx.x * BN;

    const __bf16* Ag = A + (long long)m0 * lda;   // uniform tile base (SGPR)
    const __bf16* Bg = B + n0;                    // uniform tile base (SGPR)

    const unsigned asBase[2] = { lds_off(&As[0][0]), lds_off(&As[1][0]) };
    const unsigned bsBase[2] = { lds_off(&Bs[0][0]), lds_off(&Bs[1][0]) };

    // Per-thread staging coordinates (constant across K-steps).
    // A tile: element e = (tid + v*256)*8 -> (row, col) in BM x 32.
    // B tile: element e = (tid + v*256)*8 -> (row, col) in 32 x BN.
    const unsigned long long aBase64 = (unsigned long long)Ag;
    const unsigned long long bBase64 = (unsigned long long)Bg;

    // Issue one K-tile's async global->LDS copies (16B per lane per issue).
    auto issue_tile = [&](int k0, int buf) {
        #pragma unroll
        for (int v = 0; v < AVECS; ++v) {
            const int e = (tid + v * 256) * 8;
            const int r = e / KT, c = e % KT;
            const unsigned ldsA = asBase[buf] + (unsigned)((r * ASTR + c) * 2);
            const unsigned ga   = (unsigned)(((long long)r * lda + k0 + c) * 2);
            asm volatile("global_load_async_to_lds_b128 %0, %1, %2"
                         :: "v"(ldsA), "v"(ga), "s"(aBase64) : "memory");
        }
        #pragma unroll
        for (int v = 0; v < BVECS; ++v) {
            const int e = (tid + v * 256) * 8;
            const int r = e / BN, c = e % BN;
            const unsigned ldsB = bsBase[buf] + (unsigned)((r * BSTR + c) * 2);
            const unsigned gb   = (unsigned)(((long long)(k0 + r) * ldb + c) * 2);
            asm volatile("global_load_async_to_lds_b128 %0, %1, %2"
                         :: "v"(ldsB), "v"(gb), "s"(bBase64) : "memory");
        }
    };

    v8f acc[MF][NF];
    #pragma unroll
    for (int i = 0; i < MF; ++i)
        #pragma unroll
        for (int j = 0; j < NF; ++j)
            #pragma unroll
            for (int q = 0; q < 8; ++q) acc[i][j][q] = 0.f;

    const int nk = K / KT;
    issue_tile(0, 0);

    for (int t = 0; t < nk; ++t) {
        const int buf = t & 1;
        // Wait for this wave's async copies of tile t, then workgroup barrier
        // so every wave's share of the tile is visible in LDS.
        asm volatile("s_wait_asynccnt 0x0" ::: "memory");
        __syncthreads();

        // Prefetch tile t+1 into the other buffer while we compute.
        if (t + 1 < nk) issue_tile((t + 1) * KT, buf ^ 1);

        const __bf16* asb = As[buf];
        const __bf16* bsb = Bs[buf];

        FragU af[MF];
        #pragma unroll
        for (int im = 0; im < MF; ++im) {
            const __bf16* p =
                &asb[(wm * WM + im * 16 + (lane & 15)) * ASTR + ((lane >> 4) << 3)];
            af[im].q[0] = *(const uint4*)p;
            af[im].q[1] = *(const uint4*)(p + 16);
        }
        FragU bf_[NF];
        #pragma unroll
        for (int jn = 0; jn < NF; ++jn) {
            const __bf16* p = &bsb[((lane >> 4) << 3) * BSTR + wn * WN + jn * 16 + (lane & 15)];
            #pragma unroll
            for (int s = 0; s < 8; ++s) {
                bf_[jn].v[s]     = p[s * BSTR];
                bf_[jn].v[8 + s] = p[(16 + s) * BSTR];
            }
        }
        #pragma unroll
        for (int im = 0; im < MF; ++im)
            #pragma unroll
            for (int jn = 0; jn < NF; ++jn)
                acc[im][jn] = __builtin_amdgcn_wmma_f32_16x16x32_bf16(
                    false, af[im].v, false, bf_[jn].v,
                    (short)0, acc[im][jn], false, false);

        // All waves done reading buf before the next iteration's async writes
        // can land in buf (they target buf^1 now, buf next time).
        __syncthreads();
    }

    // Epilogue. C/D layout: VGPR i -> M = i + (lane>=16 ? 8 : 0), N = lane&15.
    const int cr = ((lane >> 4) & 1) * 8;
    const int cc = lane & 15;
    #pragma unroll
    for (int im = 0; im < MF; ++im) {
        #pragma unroll
        for (int jn = 0; jn < NF; ++jn) {
            #pragma unroll
            for (int i = 0; i < 8; ++i) {
                const int gr = m0 + wm * WM + im * 16 + cr + i;
                const int gc = n0 + wn * WN + jn * 16 + cc;
                float val = acc[im][jn][i];
                const long long idx = cOff + (long long)gr * ldc + gc;
                if constexpr (EPI == EPI_BF16) {
                    Cb[idx] = (__bf16)val;
                } else if constexpr (EPI == EPI_SCORES) {
                    val *= scale;
                    if (gc > gr) val = -1e9f;      // causal mask fill
                    Cf[idx] = val;                  // fp32 scores -> softmax
                    Cb[idx] = (__bf16)val;          // bf16 scores -> a@V
                } else if constexpr (EPI == EPI_BIAS_F32) {
                    Cf[idx] = val + bias[gc];
                } else if constexpr (EPI == EPI_GELU_BF16) {
                    val += bias[gc];
                    val = 0.5f * val * (1.f + erff(val * 0.70710678118654752f));
                    Cb[idx] = (__bf16)val;
                }
            }
        }
    }
}

// ---------------------------------------------------------------------------
// Elementwise fp32 -> bf16
// ---------------------------------------------------------------------------
__global__ void cvt_bf16_kernel(const float* __restrict__ in,
                                __bf16* __restrict__ out, long long n)
{
    long long i = (long long)blockIdx.x * 256 + threadIdx.x;
    if (i < n) out[i] = (__bf16)in[i];
}

// fp32 [N,K] row-major  ->  bf16 [K,N] row-major (weight transpose to K-major)
__global__ void cvt_transpose_kernel(const float* __restrict__ in,
                                     __bf16* __restrict__ out, int N, int K)
{
    long long i = (long long)blockIdx.x * 256 + threadIdx.x;
    if (i >= (long long)N * K) return;
    int k = (int)(i % K);
    int n = (int)(i / K);
    out[(long long)k * N + n] = (__bf16)in[i];
}

// K projection [B*S, D] bf16 -> per-head K-major Kt[b,h,d,t]
__global__ void head_transpose_kernel(const __bf16* __restrict__ in,
                                      __bf16* __restrict__ out)
{
    long long i = (long long)blockIdx.x * 256 + threadIdx.x;
    if (i >= (long long)BSc * Dc) return;
    int t = (int)(i % Sc);
    long long r = i / Sc;
    int d = (int)(r % DHc);
    long long r2 = r / DHc;
    int h = (int)(r2 % Hc);
    int b = (int)(r2 / Hc);
    out[i] = in[((long long)b * Sc + t) * Dc + h * DHc + d];
}

// ---------------------------------------------------------------------------
// In-place row softmax over rows of length 1024 (one block per row).
// ---------------------------------------------------------------------------
__global__ __launch_bounds__(256) void softmax_kernel(float* __restrict__ P)
{
    constexpr int N = Sc;
    float* p = P + (long long)blockIdx.x * N;
    const int tid = threadIdx.x, lane = tid & 31, wave = tid >> 5;
    __shared__ float red[8];

    float v[4];
    float mx = -3.4e38f;
    #pragma unroll
    for (int i = 0; i < 4; ++i) { v[i] = p[tid + i * 256]; mx = fmaxf(mx, v[i]); }
    #pragma unroll
    for (int o = 16; o > 0; o >>= 1) mx = fmaxf(mx, __shfl_xor(mx, o, 32));
    if (lane == 0) red[wave] = mx;
    __syncthreads();
    float rmx = red[0];
    #pragma unroll
    for (int w = 1; w < 8; ++w) rmx = fmaxf(rmx, red[w]);
    __syncthreads();

    float s = 0.f;
    #pragma unroll
    for (int i = 0; i < 4; ++i) { v[i] = __expf(v[i] - rmx); s += v[i]; }
    #pragma unroll
    for (int o = 16; o > 0; o >>= 1) s += __shfl_xor(s, o, 32);
    if (lane == 0) red[wave] = s;
    __syncthreads();
    float tot = 0.f;
    #pragma unroll
    for (int w = 0; w < 8; ++w) tot += red[w];
    const float inv = 1.f / tot;
    #pragma unroll
    for (int i = 0; i < 4; ++i) p[tid + i * 256] = v[i] * inv;
}

// ---------------------------------------------------------------------------
// h = LayerNorm(x + attn_out) * g + b   -> bf16 (one block per row of 1024)
// ---------------------------------------------------------------------------
__global__ __launch_bounds__(256) void add_ln_kernel(
    const float* __restrict__ x, const float* __restrict__ r,
    const float* __restrict__ g, const float* __restrict__ bt,
    __bf16* __restrict__ out)
{
    constexpr int D = Dc;
    const long long row = blockIdx.x;
    const int tid = threadIdx.x, lane = tid & 31, wave = tid >> 5;
    __shared__ float red[8];
    const float* xr = x + row * D;
    const float* rr = r + row * D;

    float v[4];
    float s = 0.f;
    #pragma unroll
    for (int i = 0; i < 4; ++i) {
        int e = tid + i * 256;
        v[i] = xr[e] + rr[e];
        s += v[i];
    }
    #pragma unroll
    for (int o = 16; o > 0; o >>= 1) s += __shfl_xor(s, o, 32);
    if (lane == 0) red[wave] = s;
    __syncthreads();
    float tot = 0.f;
    #pragma unroll
    for (int w = 0; w < 8; ++w) tot += red[w];
    const float mean = tot * (1.f / D);
    __syncthreads();

    float vs = 0.f;
    #pragma unroll
    for (int i = 0; i < 4; ++i) { float d = v[i] - mean; vs += d * d; }
    #pragma unroll
    for (int o = 16; o > 0; o >>= 1) vs += __shfl_xor(vs, o, 32);
    if (lane == 0) red[wave] = vs;
    __syncthreads();
    float vtot = 0.f;
    #pragma unroll
    for (int w = 0; w < 8; ++w) vtot += red[w];
    const float inv = rsqrtf(vtot * (1.f / D) + 1e-5f);
    #pragma unroll
    for (int i = 0; i < 4; ++i) {
        int e = tid + i * 256;
        out[row * D + e] = (__bf16)((v[i] - mean) * inv * g[e] + bt[e]);
    }
}

// ---------------------------------------------------------------------------
// Launch
// ---------------------------------------------------------------------------
extern "C" void kernel_launch(void* const* d_in, const int* in_sizes, int n_in,
                              void* d_out, int out_size, void* d_ws, size_t ws_size,
                              hipStream_t stream)
{
    (void)in_sizes; (void)n_in; (void)out_size; (void)ws_size;
    const float* x    = (const float*)d_in[0];
    // d_in[1] = attn_mask (causal, computed analytically)
    const float* Wq   = (const float*)d_in[2];
    const float* Wk   = (const float*)d_in[3];
    const float* Wv   = (const float*)d_in[4];
    const float* Wo   = (const float*)d_in[5];
    const float* bo   = (const float*)d_in[6];
    const float* ln1g = (const float*)d_in[7];
    const float* ln1b = (const float*)d_in[8];
    const float* w1   = (const float*)d_in[9];
    const float* b1   = (const float*)d_in[10];
    const float* w2   = (const float*)d_in[11];
    const float* b2   = (const float*)d_in[12];

    float* out    = (float*)d_out;                       // ffn_out [B,S,D]
    float* attn_p = out + (long long)BSc * Dc;           // attn_p [B,H,S,S]

    // Workspace carve-out (256B aligned)
    char* ws = (char*)d_ws;
    size_t off = 0;
    auto alloc = [&](size_t bytes) -> void* {
        off = (off + 255) & ~(size_t)255;
        void* p = ws + off;
        off += bytes;
        return p;
    };
    __bf16* xb   = (__bf16*)alloc((size_t)BSc * Dc * 2);
    __bf16* Wqt  = (__bf16*)alloc((size_t)Dc * Dc * 2);
    __bf16* Wkt  = (__bf16*)alloc((size_t)Dc * Dc * 2);
    __bf16* Wvt  = (__bf16*)alloc((size_t)Dc * Dc * 2);
    __bf16* Wot  = (__bf16*)alloc((size_t)Dc * Dc * 2);
    __bf16* w1t  = (__bf16*)alloc((size_t)Dc * DFFc * 2);
    __bf16* w2t  = (__bf16*)alloc((size_t)DFFc * Dc * 2);
    __bf16* Qb   = (__bf16*)alloc((size_t)BSc * Dc * 2);
    __bf16* Kb   = (__bf16*)alloc((size_t)BSc * Dc * 2);
    __bf16* Vb   = (__bf16*)alloc((size_t)BSc * Dc * 2);
    __bf16* Ktb  = (__bf16*)alloc((size_t)BSc * Dc * 2);
    __bf16* abf  = (__bf16*)alloc((size_t)Bc * Hc * Sc * Sc * 2);  // 128 MB
    __bf16* avb  = (__bf16*)alloc((size_t)BSc * Dc * 2);
    float*  aout = (float*)alloc((size_t)BSc * Dc * 4);
    __bf16* hb   = (__bf16*)alloc((size_t)BSc * Dc * 2);
    __bf16* hhb  = (__bf16*)alloc((size_t)BSc * DFFc * 2);

    const long long nXD = (long long)BSc * Dc;

    // 1) Precision conversion (+ weight transposes to K-major)
    cvt_bf16_kernel<<<dim3((unsigned)((nXD + 255) / 256)), 256, 0, stream>>>(x, xb, nXD);
    cvt_transpose_kernel<<<dim3((unsigned)((Dc * Dc + 255) / 256)), 256, 0, stream>>>(Wq, Wqt, Dc, Dc);
    cvt_transpose_kernel<<<dim3((unsigned)((Dc * Dc + 255) / 256)), 256, 0, stream>>>(Wk, Wkt, Dc, Dc);
    cvt_transpose_kernel<<<dim3((unsigned)((Dc * Dc + 255) / 256)), 256, 0, stream>>>(Wv, Wvt, Dc, Dc);
    cvt_transpose_kernel<<<dim3((unsigned)((Dc * Dc + 255) / 256)), 256, 0, stream>>>(Wo, Wot, Dc, Dc);
    cvt_transpose_kernel<<<dim3((unsigned)((DFFc * Dc + 255) / 256)), 256, 0, stream>>>(w1, w1t, DFFc, Dc);
    cvt_transpose_kernel<<<dim3((unsigned)((Dc * DFFc + 255) / 256)), 256, 0, stream>>>(w2, w2t, Dc, DFFc);

    // 2) Q/K/V projections: [4096x1024] = xb @ Wt
    dim3 gProj(Dc / 128, BSc / 128, 1);
    gemm_bf16_kernel<128,128,32,64,EPI_BF16><<<gProj, 256, 0, stream>>>(
        xb, Wqt, nullptr, Qb, Dc, Dc, Dc, Dc, 1, 0,0, 0,0, 0,0, nullptr, 1.f);
    gemm_bf16_kernel<128,128,32,64,EPI_BF16><<<gProj, 256, 0, stream>>>(
        xb, Wkt, nullptr, Kb, Dc, Dc, Dc, Dc, 1, 0,0, 0,0, 0,0, nullptr, 1.f);
    gemm_bf16_kernel<128,128,32,64,EPI_BF16><<<gProj, 256, 0, stream>>>(
        xb, Wvt, nullptr, Vb, Dc, Dc, Dc, Dc, 1, 0,0, 0,0, 0,0, nullptr, 1.f);

    // 3) Per-head K transpose -> Kt[b,h,d,t] (K-major for the scores GEMM)
    head_transpose_kernel<<<dim3((unsigned)((nXD + 255) / 256)), 256, 0, stream>>>(Kb, Ktb);

    // 4) Scores: a = (q @ k^T)/8, causal mask. fp32 -> attn_p region, bf16 -> abf
    gemm_bf16_kernel<128,128,32,64,EPI_SCORES><<<dim3(Sc / 128, Sc / 128, Bc * Hc), 256, 0, stream>>>(
        Qb, Ktb, attn_p, abf,
        Dc, Sc, Sc, DHc, Hc,
        (long long)Sc * Dc, DHc,
        (long long)Hc * DHc * Sc, (long long)DHc * Sc,
        (long long)Hc * Sc * Sc, (long long)Sc * Sc,
        nullptr, 0.125f);

    // 5) Softmax in place on attn_p (B*H*S rows)
    softmax_kernel<<<dim3(Bc * Hc * Sc), 256, 0, stream>>>(attn_p);

    // 6) attn_v = a @ v  (raw masked scores, per reference) -> [B*S, D] bf16
    gemm_bf16_kernel<128,64,32,32,EPI_BF16><<<dim3(DHc / 64, Sc / 128, Bc * Hc), 256, 0, stream>>>(
        abf, Vb, nullptr, avb,
        Sc, Dc, Dc, Sc, Hc,
        (long long)Hc * Sc * Sc, (long long)Sc * Sc,
        (long long)Sc * Dc, DHc,
        (long long)Sc * Dc, DHc,
        nullptr, 1.f);

    // 7) attn_out = attn_v @ Wo^T + bo  (fp32)
    gemm_bf16_kernel<128,128,32,64,EPI_BIAS_F32><<<gProj, 256, 0, stream>>>(
        avb, Wot, aout, nullptr, Dc, Dc, Dc, Dc, 1, 0,0, 0,0, 0,0, bo, 1.f);

    // 8) h = LN(x + attn_out) -> bf16
    add_ln_kernel<<<dim3(BSc), 256, 0, stream>>>(x, aout, ln1g, ln1b, hb);

    // 9) FFN1: gelu(h @ w1^T + b1) -> bf16 [4096x4096]
    gemm_bf16_kernel<128,128,32,64,EPI_GELU_BF16><<<dim3(DFFc / 128, BSc / 128, 1), 256, 0, stream>>>(
        hb, w1t, nullptr, hhb, Dc, DFFc, DFFc, Dc, 1, 0,0, 0,0, 0,0, b1, 1.f);

    // 10) FFN2: ffn_out = hh @ w2^T + b2 -> fp32 d_out
    gemm_bf16_kernel<128,128,32,64,EPI_BIAS_F32><<<dim3(Dc / 128, BSc / 128, 1), 256, 0, stream>>>(
        hhb, w2t, out, nullptr, DFFc, Dc, Dc, DFFc, 1, 0,0, 0,0, 0,0, b2, 1.f);
}
